// DependencyParser_78675210928657
// MI455X (gfx1250) — matrix operand order repaired and checked by reading, not compiled, not run on hardware
//
#include <hip/hip_runtime.h>
#include <math.h>

// Problem dims
#define BB 16
#define SS 128
#define IND 200
#define HH 256

typedef __attribute__((ext_vector_type(16))) _Float16 v16h;
typedef __attribute__((ext_vector_type(8)))  _Float16 v8h;
typedef __attribute__((ext_vector_type(8)))  float    v8f;

union F16x16 { v16h v; v8h p[2]; _Float16 e[16]; };
union F32x8  { v8f  v; float e[8]; };

__device__ __forceinline__ v8f wmma16x16x32(v16h a, v16h b, v8f c) {
  // D = A(16x32 f16) * B(32x16 f16) + C(16x16 f32)
  return __builtin_amdgcn_wmma_f32_16x16x32_f16(false, a, false, b, (short)0, c, false, false);
}

__device__ __forceinline__ float sigmoidf_(float x) { return 1.0f / (1.0f + expf(-x)); }
__device__ __forceinline__ float reluf_(float x) { return x > 0.0f ? x : 0.0f; }

// ---------------------------------------------------------------------------
// Generic f32 -> f16 row conversion with stride change and zero pad:
// dst[r*dld + c] = (c < cols) ? (f16)src[r*sld + c] : 0   for c in [0, dld)
// ---------------------------------------------------------------------------
__global__ void cvt_rows_f16(const float* __restrict__ src, long long sld,
                             _Float16* __restrict__ dst, int dld,
                             int rows, int cols) {
  long long total = (long long)rows * dld;
  long long i = (long long)blockIdx.x * blockDim.x + threadIdx.x;
  long long stride = (long long)gridDim.x * blockDim.x;
  for (; i < total; i += stride) {
    int r = (int)(i / dld);
    int c = (int)(i - (long long)r * dld);
    dst[i] = (c < cols) ? (_Float16)src[(long long)r * sld + c] : (_Float16)0.0f;
  }
}

// sentence (16,128,200) f32 -> time-major rows r=t*16+b, f16, padded to 224
__global__ void cvt_sentence_f16(const float* __restrict__ src, _Float16* __restrict__ dst) {
  long long total = 2048LL * 224LL;
  long long i = (long long)blockIdx.x * blockDim.x + threadIdx.x;
  long long stride = (long long)gridDim.x * blockDim.x;
  for (; i < total; i += stride) {
    int r = (int)(i / 224);
    int c = (int)(i - (long long)r * 224);
    int t = r >> 4;
    int b = r & 15;
    dst[i] = (c < 200) ? (_Float16)src[((long long)b * 128 + t) * 200 + c] : (_Float16)0.0f;
  }
}

// ---------------------------------------------------------------------------
// f16 WMMA GEMM: out[r, n] = sum_k A[r, k] * W[n, k] + bias0[n] + bias1[n]
// A: (M, lda) f16 row-major, W: (N, ldw) f16 row-major, K multiple of 32.
// Branch-free fragment loads: two contiguous 128-bit loads per operand.
// Each wave computes one 16x16 tile; 8 waves per block.
// ---------------------------------------------------------------------------
__global__ void gemm_f16_wmma(const _Float16* __restrict__ A, int lda,
                              const _Float16* __restrict__ W, int ldw,
                              int K, int N,
                              const float* __restrict__ bias0, const float* __restrict__ bias1,
                              float* __restrict__ out, int tiles_n)
{
  int lane = threadIdx.x & 31;
  int wave = threadIdx.x >> 5;
  int tile = blockIdx.x * 8 + wave;
  int tm = tile / tiles_n;
  int tn = tile % tiles_n;
  int m0 = tm * 16, n0 = tn * 16;

  int kgrp = lane >> 4;                 // 0 or 1 (lane half)
  const _Float16* arow = A + (long long)(m0 + (lane & 15)) * lda + kgrp * 8;
  const _Float16* wrow = W + (long long)(n0 + (lane & 15)) * ldw + kgrp * 8;

  F32x8 acc = {};
  for (int kk = 0; kk < K; kk += 32) {
    F16x16 a, bf;
    a.p[0]  = *(const v8h*)(arow + kk);        // k = kk + kgrp*8 + [0,8)
    a.p[1]  = *(const v8h*)(arow + kk + 16);   // k = kk + 16 + kgrp*8 + [0,8)
    bf.p[0] = *(const v8h*)(wrow + kk);        // k = kk + kgrp*8 + [0,8)
    bf.p[1] = *(const v8h*)(wrow + kk + 16);   // k = kk + 16 + kgrp*8 + [0,8)
    acc.v = wmma16x16x32(a.v, bf.v, acc.v);
  }
  int n = n0 + (lane & 15);
  float bias = (bias0 ? bias0[n] : 0.0f) + (bias1 ? bias1[n] : 0.0f);
#pragma unroll
  for (int rr = 0; rr < 8; ++rr) {
    int m = m0 + rr + kgrp * 8;                // ISA C/D layout
    out[(long long)m * N + n] = acc.e[rr] + bias;
  }
}

// ---------------------------------------------------------------------------
// Persistent single-workgroup LSTM recurrence for one direction.
// xw: (S,16,1024) f32 precomputed x@Wih^T + bih + bhh (natural time order)
// whh: (1024,256) f16   emb16: (S,16,512) f16, writes columns [out_off, out_off+256)
// blockDim = 1024 (32 waves); each wave owns 2 N-tiles of the 16x1024 gate GEMM.
// ---------------------------------------------------------------------------
__global__ void lstm_recurrent(const float* __restrict__ xw,
                               const _Float16* __restrict__ whh,
                               _Float16* __restrict__ emb16,
                               int out_off, int reverse)
{
  extern __shared__ char smem[];
  float*    c_s = (float*)smem;                                 // 16*256 f32  (16 KB)
  _Float16* h16 = (_Float16*)(smem + 16 * 256 * 4);             // 16*256 f16  (8 KB)
  float*    g_s = (float*)(smem + 16 * 256 * 4 + 16 * 256 * 2); // 16*1024 f32 (64 KB)

  int tid = threadIdx.x;
  int lane = tid & 31;
  int wave = tid >> 5;
  int kgrp = lane >> 4;

  for (int i = tid; i < 16 * 256; i += 1024) { c_s[i] = 0.0f; h16[i] = (_Float16)0.0f; }
  __syncthreads();

  for (int step = 0; step < SS; ++step) {
    int t = reverse ? (SS - 1 - step) : step;
    const float* xwrow = xw + (long long)t * (16 * 1024);

    // g = h @ Whh^T : M=16, N=1024, K=256
#pragma unroll
    for (int rep = 0; rep < 2; ++rep) {
      int n0 = (wave * 2 + rep) * 16;
      const _Float16* arow = h16 + (lane & 15) * 256 + kgrp * 8;
      const _Float16* wrow = whh + (long long)(n0 + (lane & 15)) * 256 + kgrp * 8;
      F32x8 acc = {};
      for (int kk = 0; kk < 256; kk += 32) {
        F16x16 a, bm;
        a.p[0]  = *(const v8h*)(arow + kk);
        a.p[1]  = *(const v8h*)(arow + kk + 16);
        bm.p[0] = *(const v8h*)(wrow + kk);
        bm.p[1] = *(const v8h*)(wrow + kk + 16);
        acc.v = wmma16x16x32(a.v, bm.v, acc.v);
      }
      int n = n0 + (lane & 15);
#pragma unroll
      for (int rr = 0; rr < 8; ++rr) {
        int m = rr + kgrp * 8;
        g_s[m * 1024 + n] = acc.e[rr] + xwrow[m * 1024 + n];
      }
    }
    __syncthreads();

    // gates: 16 batches x 256 hidden = 4096 elems / 1024 threads
#pragma unroll
    for (int u = 0; u < 4; ++u) {
      int idx = tid + u * 1024;
      int b = idx >> 8;
      int j = idx & 255;
      float gi = g_s[b * 1024 + j];
      float gf = g_s[b * 1024 + j + 256];
      float gg = g_s[b * 1024 + j + 512];
      float go = g_s[b * 1024 + j + 768];
      float c = sigmoidf_(gf) * c_s[b * 256 + j] + sigmoidf_(gi) * tanhf(gg);
      float h = sigmoidf_(go) * tanhf(c);
      c_s[b * 256 + j] = c;
      _Float16 h_16 = (_Float16)h;
      h16[b * 256 + j] = h_16;
      emb16[((long long)t * 16 + b) * 512 + out_off + j] = h_16;
    }
    __syncthreads();
  }
}

// ---------------------------------------------------------------------------
// Fused pairwise scorer: one workgroup per (b, i).
// h1[j,:] = relu(Ha[i]+Hb[j]+b1) kept in LDS as f16 (128x256);
// h2 = relu(h1 @ W2^T + b2) via WMMA (M=128,K=256,N=128) into LDS f32;
// score[j] = relu(h2[j]·W3 + b3) * mask.
// ---------------------------------------------------------------------------
__global__ void score_kernel(const float* __restrict__ Ha, const float* __restrict__ Hb,
                             const float* __restrict__ b1,
                             const _Float16* __restrict__ W2h, const float* __restrict__ b2,
                             const float* __restrict__ W3, const float* __restrict__ b3v,
                             float* __restrict__ d_out)
{
  extern __shared__ char smem[];
  _Float16* h1     = (_Float16*)smem;                                  // 128*256 f16 (64 KB)
  float*    h2     = (float*)(smem + 128 * 256 * 2);                   // 128*128 f32 (64 KB)
  float*    ha_row = (float*)(smem + 128 * 256 * 2 + 128 * 128 * 4);   // 256 f32

  int tid = threadIdx.x;           // blockDim = 256
  int bIdx = blockIdx.x >> 7;      // batch
  int i = blockIdx.x & 127;        // head-position row

  ha_row[tid] = Ha[((long long)i * 16 + bIdx) * 256 + tid];
  __syncthreads();

  // build h1: thread owns one feature d across all j (coalesced Hb reads)
  {
    int d = tid;
    float hav = ha_row[d] + b1[d];
    for (int j = 0; j < 128; ++j) {
      float v = hav + Hb[((long long)j * 16 + bIdx) * 256 + d];
      h1[j * 256 + d] = (_Float16)reluf_(v);
    }
  }
  __syncthreads();

  // h2 = relu(h1 @ W2^T + b2): wave w owns m-tile w, loops 8 n-tiles, K=256
  {
    int lane = tid & 31;
    int wave = tid >> 5;
    int kgrp = lane >> 4;
    int m0 = wave * 16;
    const _Float16* arow = h1 + (m0 + (lane & 15)) * 256 + kgrp * 8;
    for (int tn = 0; tn < 8; ++tn) {
      int n0 = tn * 16;
      const _Float16* wrow = W2h + (long long)(n0 + (lane & 15)) * 256 + kgrp * 8;
      F32x8 acc = {};
      for (int kk = 0; kk < 256; kk += 32) {
        F16x16 a, bm;
        a.p[0]  = *(const v8h*)(arow + kk);
        a.p[1]  = *(const v8h*)(arow + kk + 16);
        bm.p[0] = *(const v8h*)(wrow + kk);
        bm.p[1] = *(const v8h*)(wrow + kk + 16);
        acc.v = wmma16x16x32(a.v, bm.v, acc.v);
      }
      int n = n0 + (lane & 15);
      float bb = b2[n];
#pragma unroll
      for (int rr = 0; rr < 8; ++rr) {
        int m = m0 + rr + kgrp * 8;
        h2[m * 128 + n] = reluf_(acc.e[rr] + bb);
      }
    }
  }
  __syncthreads();

  // score row: dot with W3, relu, mask, store
  if (tid < 128) {
    int j = tid;
    float s = b3v[0];
    for (int n = 0; n < 128; ++n) s += h2[j * 128 + n] * W3[n];
    s = reluf_(s);
    if (j == i || j < 1) s = 0.0f;
    d_out[1 + ((long long)blockIdx.x * 128 + j)] = s;
  }
}

// ---------------------------------------------------------------------------
// Loss: per-batch partial (fixed-order tree reduce, deterministic), then sum.
// ---------------------------------------------------------------------------
__global__ void loss_partial_kernel(const float* __restrict__ score,  // (16,128,128), masked
                                    const int* __restrict__ tree,     // (16,128,2)
                                    float* __restrict__ partial)
{
  __shared__ float red[128];
  int b = blockIdx.x;
  int k = threadIdx.x;
  float val = 0.0f;
  if (k < 127) {
    int head = tree[(b * 128 + (k + 1)) * 2 + 0];
    int dep  = tree[(b * 128 + (k + 1)) * 2 + 1];
    float norm = 0.0f;
    for (int ii = 0; ii < 128; ++ii) {
      if (ii != dep) norm += expf(score[((long long)b * 128 + ii) * 128 + dep]);
    }
    float num = score[((long long)b * 128 + head) * 128 + dep];
    val = logf(norm) - num;
  }
  red[k] = val;
  __syncthreads();
  for (int s = 64; s > 0; s >>= 1) {
    if (k < s) red[k] += red[k + s];
    __syncthreads();
  }
  if (k == 0) partial[b] = red[0] / 127.0f;
}

__global__ void loss_final_kernel(const float* __restrict__ partial, float* __restrict__ d_out) {
  if (threadIdx.x == 0 && blockIdx.x == 0) {
    float s = 0.0f;
    for (int b = 0; b < 16; ++b) s += partial[b];
    d_out[0] = s;
  }
}

// ---------------------------------------------------------------------------
extern "C" void kernel_launch(void* const* d_in, const int* in_sizes, int n_in,
                              void* d_out, int out_size, void* d_ws, size_t ws_size,
                              hipStream_t stream) {
  (void)in_sizes; (void)n_in; (void)out_size; (void)ws_size;

  const float* sentence = (const float*)d_in[0];
  const float* Wih0 = (const float*)d_in[1];
  const float* Whh0 = (const float*)d_in[2];
  const float* bih0 = (const float*)d_in[3];
  const float* bhh0 = (const float*)d_in[4];
  const float* Wih1 = (const float*)d_in[5];
  const float* Whh1 = (const float*)d_in[6];
  const float* bih1 = (const float*)d_in[7];
  const float* bhh1 = (const float*)d_in[8];
  const float* W1 = (const float*)d_in[9];
  const float* b1 = (const float*)d_in[10];
  const float* W2 = (const float*)d_in[11];
  const float* b2 = (const float*)d_in[12];
  const float* W3 = (const float*)d_in[13];
  const float* b3 = (const float*)d_in[14];
  const int*   tree = (const int*)d_in[15];
  float* out = (float*)d_out;

  // Workspace layout (bytes, 64-aligned blocks), total ~31.8 MB
  char* ws = (char*)d_ws;
  _Float16* whh16   = (_Float16*)(ws + 0);          // 4 x (1024x256) f16
  _Float16* w2h     = (_Float16*)(ws + 2097152);    // (128x256) f16
  _Float16* x16     = (_Float16*)(ws + 2162688);    // (2048x224) f16, time-major, padded
  _Float16* wih0_16 = (_Float16*)(ws + 3080192);    // 2 x (1024x224) f16, padded
  _Float16* wih1_16 = (_Float16*)(ws + 3997696);    // 2 x (1024x512) f16
  _Float16* w1a16   = (_Float16*)(ws + 6094848);    // (256x512) f16  (W1[:, :512])
  _Float16* w1b16   = (_Float16*)(ws + 6356992);    // (256x512) f16  (W1[:, 512:])
  float* xwf   = (float*)(ws + 6619136);            // (S,16,1024) f32
  float* xwb   = (float*)(ws + 15007744);           // (S,16,1024) f32
  _Float16* emb1_16 = (_Float16*)(ws + 23396352);   // (S,16,512) f16
  _Float16* emb2_16 = (_Float16*)(ws + 25493504);   // (S,16,512) f16
  float* Ha    = (float*)(ws + 27590656);           // (S,16,256) f32
  float* Hb    = (float*)(ws + 29687808);           // (S,16,256) f32
  float* partial = (float*)(ws + 31784960);         // 16 f32

  // --- one-time f16 staging (padded where K % 32 != 0) ---
  cvt_rows_f16<<<512, 256, 0, stream>>>(Whh0, 256LL, whh16, 256, 2048, 256);
  cvt_rows_f16<<<512, 256, 0, stream>>>(Whh1, 256LL, whh16 + 2 * 262144, 256, 2048, 256);
  cvt_rows_f16<<<64, 256, 0, stream>>>(W2, 256LL, w2h, 256, 128, 256);
  cvt_sentence_f16<<<512, 256, 0, stream>>>(sentence, x16);
  cvt_rows_f16<<<512, 256, 0, stream>>>(Wih0, 200LL, wih0_16, 224, 2048, 200);
  cvt_rows_f16<<<512, 256, 0, stream>>>(Wih1, 512LL, wih1_16, 512, 2048, 512);
  cvt_rows_f16<<<256, 256, 0, stream>>>(W1, 1024LL, w1a16, 512, 256, 512);
  cvt_rows_f16<<<256, 256, 0, stream>>>(W1 + 512, 1024LL, w1b16, 512, 256, 512);

  // --- layer 0 input projections (biases folded in), K padded to 224 ---
  gemm_f16_wmma<<<1024, 256, 0, stream>>>(x16, 224, wih0_16, 224, 224, 1024,
                                          bih0, bhh0, xwf, 64);
  gemm_f16_wmma<<<1024, 256, 0, stream>>>(x16, 224, wih0_16 + 1024 * 224, 224, 224, 1024,
                                          bih0 + 1024, bhh0 + 1024, xwb, 64);

  size_t lstm_smem = 16 * 256 * 4 + 16 * 256 * 2 + 16 * 1024 * 4;  // 88 KB
  lstm_recurrent<<<1, 1024, lstm_smem, stream>>>(xwf, whh16, emb1_16, 0, 0);
  lstm_recurrent<<<1, 1024, lstm_smem, stream>>>(xwb, whh16 + 262144, emb1_16, 256, 1);

  // --- layer 1 input projections from emb1 (f16, time-major (s,b,512)) ---
  gemm_f16_wmma<<<1024, 256, 0, stream>>>(emb1_16, 512, wih1_16, 512, 512, 1024,
                                          bih1, bhh1, xwf, 64);
  gemm_f16_wmma<<<1024, 256, 0, stream>>>(emb1_16, 512, wih1_16 + 1024 * 512, 512, 512, 1024,
                                          bih1 + 1024, bhh1 + 1024, xwb, 64);

  lstm_recurrent<<<1, 1024, lstm_smem, stream>>>(xwf, whh16 + 2 * 262144, emb2_16, 0, 0);
  lstm_recurrent<<<1, 1024, lstm_smem, stream>>>(xwb, whh16 + 3 * 262144, emb2_16, 256, 1);

  // --- Ha = emb2 @ W1[:, :512]^T ; Hb = emb2 @ W1[:, 512:]^T ---
  gemm_f16_wmma<<<256, 256, 0, stream>>>(emb2_16, 512, w1a16, 512, 512, 256,
                                         nullptr, nullptr, Ha, 16);
  gemm_f16_wmma<<<256, 256, 0, stream>>>(emb2_16, 512, w1b16, 512, 512, 256,
                                         nullptr, nullptr, Hb, 16);

  // --- fused pairwise MLP + score (dominant compute, all in LDS) ---
  size_t score_smem = 128 * 256 * 2 + 128 * 128 * 4 + 256 * 4;  // ~129 KB
  score_kernel<<<2048, 256, score_smem, stream>>>(Ha, Hb, b1, w2h, b2, W3, b3, out);

  // --- loss ---
  loss_partial_kernel<<<16, 128, 0, stream>>>(out + 1, tree, partial);
  loss_final_kernel<<<1, 32, 0, stream>>>(partial, out);
}
